// LightweightAttention_33852932227358
// MI455X (gfx1250) — matrix operand order repaired and checked by reading, not compiled, not run on hardware
//
#include <hip/hip_runtime.h>
#include <math.h>

// ---------------------------------------------------------------------------
// Linear attention (elu+1 feature map) for MI455X / gfx1250.
// All matmuls via v_wmma_f32_16x16x32_bf16 (bf16 inputs, fp32 accumulate).
// Round 3: big GEMMs stage the shared 64x32 weight tile in LDS (double
// buffered, bank-conflict-free padded layout) so the 8 waves of a WG stop
// loading identical B fragments 8x from global; barrier-pipelined k-loop.
// ---------------------------------------------------------------------------

typedef __attribute__((ext_vector_type(16))) __bf16       bhalf16;
typedef __attribute__((ext_vector_type(8)))  float        floatx8;
typedef __attribute__((ext_vector_type(4))) unsigned int  uintx4;

union FragU {
  bhalf16 bf;
  uintx4  u4[2];
  __bf16  e[16];
};

__device__ __forceinline__ floatx8 zero8() {
  floatx8 z = {0.f, 0.f, 0.f, 0.f, 0.f, 0.f, 0.f, 0.f};
  return z;
}

// A fragment (16x32 bf16, M x K) direct from global: lane lr holds row M=lr.
// elements 0..15 = K in [k0 + half*16, k0 + half*16 + 15]
__device__ __forceinline__ bhalf16 load_fragA_row(const __bf16* rowbase, int ld,
                                                  int k0, int lr, int half) {
  const __bf16* p = rowbase + (size_t)lr * ld + k0 + half * 16;
  FragU f;
  f.u4[0] = *(const uintx4*)p;
  f.u4[1] = *(const uintx4*)(p + 8);
  return f.bf;
}

// B fragment (32x16 bf16, K x N) from a [N][K] row-major matrix in global
// (torch Linear weight layout): lane lr holds column N=lr; 16 contiguous K
// values starting at k0 + half*16.
__device__ __forceinline__ bhalf16 load_fragB_wt(const __bf16* wbase, int ld,
                                                 int lr, int half, int k0) {
  const __bf16* p = wbase + (size_t)lr * ld + k0 + half * 16;
  FragU f;
  f.u4[0] = *(const uintx4*)p;
  f.u4[1] = *(const uintx4*)(p + 8);
  return f.bf;
}

// B fragment from an LDS tile laid out [n][k], n-local 0..63, k-local 0..31,
// row stride TSTR=40 elements (80 B = 20 banks -> conflict-free b128 reads).
#define TSTR 40
__device__ __forceinline__ bhalf16 lds_fragB(const unsigned short* tile,
                                             int nloc, int half) {
  const unsigned short* p = tile + nloc * TSTR + half * 16;
  FragU f;
  f.u4[0] = *(const uintx4*)p;
  f.u4[1] = *(const uintx4*)(p + 8);
  return f.bf;
}

__device__ __forceinline__ floatx8 wmma_bf16(bhalf16 a, bhalf16 b, floatx8 c) {
  return __builtin_amdgcn_wmma_f32_16x16x32_bf16(false, a, false, b,
                                                 (short)0, c, false, false);
}

// ---------------------------------------------------------------------------
// fp32 -> bf16 conversion
// ---------------------------------------------------------------------------
__global__ void la_cvt_bf16(const float* __restrict__ src,
                            unsigned short* __restrict__ dst, int n) {
  int i = (blockIdx.x * blockDim.x + threadIdx.x) * 2;
  if (i + 1 < n) {
    __bf16* d = (__bf16*)dst;
    d[i]     = (__bf16)src[i];
    d[i + 1] = (__bf16)src[i + 1];
  }
}

// ---------------------------------------------------------------------------
// Fused Q/K/V projection GEMM: out = act(x @ W^T + b), z selects Q/K/V.
// M=16384, N=1024, K=1024. WG = 256 thr (8 waves); WG tile 128x64;
// wave tile 16x64. Shared B tile staged in LDS, double buffered.
// ---------------------------------------------------------------------------
__global__ __launch_bounds__(256) void la_gemm_qkv(
    const unsigned short* __restrict__ xb_raw,
    const unsigned short* __restrict__ wb_raw,
    const float* __restrict__ bq, const float* __restrict__ bk,
    const float* __restrict__ bv,
    unsigned short* __restrict__ qb_raw, unsigned short* __restrict__ kb_raw,
    unsigned short* __restrict__ vb_raw) {
  __shared__ __align__(16) unsigned short btile[2][64 * TSTR];  // 10 KB

  const __bf16* xb = (const __bf16*)xb_raw;
  const int z = blockIdx.z;
  const __bf16* W = (const __bf16*)wb_raw + (size_t)z * (1024 * 1024);
  const float* bias = (z == 0) ? bq : (z == 1) ? bk : bv;
  __bf16* out = (__bf16*)((z == 0) ? qb_raw : (z == 1) ? kb_raw : vb_raw);

  const int wave = threadIdx.x >> 5, lane = threadIdx.x & 31;
  const int lr = lane & 15, half = lane >> 4;
  const int m0 = blockIdx.x * 128 + wave * 16;
  const int n0 = blockIdx.y * 64;

  // staging role: thread -> row n=tid>>2 of the tile, 8-element k chunk
  const int sn = threadIdx.x >> 2, sk = (threadIdx.x & 3) * 8;
  const __bf16* Wst = W + (size_t)(n0 + sn) * 1024 + sk;

  const __bf16* Arow = xb + (size_t)m0 * 1024;
  floatx8 acc[4];
#pragma unroll
  for (int i = 0; i < 4; i++) acc[i] = zero8();

  // prologue: stage chunk 0 into buffer 0
  *(uintx4*)&btile[0][sn * TSTR + sk] = *(const uintx4*)Wst;

  for (int it = 0; it < 32; ++it) {
    const int k0 = it * 32;
    const int cur = it & 1;
    // issue next tile's global load early (lands in a register, stored after
    // the wmmas -> global latency overlapped with compute)
    uintx4 wreg;
    if (it + 1 < 32) wreg = *(const uintx4*)(Wst + k0 + 32);
    __syncthreads();  // staged buffer `cur` ready; buffer `cur^1` drained
    bhalf16 a = load_fragA_row(Arow, 1024, k0, lr, half);
    bhalf16 bfr[4];
#pragma unroll
    for (int ni = 0; ni < 4; ni++)
      bfr[ni] = lds_fragB(&btile[cur][0], ni * 16 + lr, half);
#pragma unroll
    for (int ni = 0; ni < 4; ni++)
      acc[ni] = wmma_bf16(a, bfr[ni], acc[ni]);
    if (it + 1 < 32) *(uintx4*)&btile[cur ^ 1][sn * TSTR + sk] = wreg;
  }

#pragma unroll
  for (int ni = 0; ni < 4; ni++) {
    const int col = n0 + ni * 16 + lr;
    const float bb = bias[col];
#pragma unroll
    for (int r = 0; r < 8; r++) {
      float v = acc[ni][r] + bb;
      if (z < 2) v = (v > 0.f) ? (v + 1.f) : expf(v);  // elu(v)+1
      out[(size_t)(m0 + r + half * 8) * 1024 + col] = (__bf16)v;
    }
  }
}

// ---------------------------------------------------------------------------
// KV aggregation stage 1 (slabbed): per (b,h,slab of 512 t)
//   partial KV^T[e][d] (fp32) and partial Ksum[d] (fp32, via WMMA vs ones).
// grid = (64 bh, 8 slabs), WG = 128 thr (4 waves); wave owns 16 rows of d.
// ---------------------------------------------------------------------------
__global__ __launch_bounds__(128) void la_kv_partial(
    const unsigned short* __restrict__ kb_raw,
    const unsigned short* __restrict__ vb_raw,
    float* __restrict__ kvp, float* __restrict__ ksump) {
  const __bf16* Kb = (const __bf16*)kb_raw;
  const __bf16* Vb = (const __bf16*)vb_raw;

  const int bh = blockIdx.x, b = bh >> 4, h = bh & 15;
  const int s = blockIdx.y;  // t-slab
  const int wave = threadIdx.x >> 5, lane = threadIdx.x & 31;
  const int lr = lane & 15, half = lane >> 4;
  const int d0 = wave * 16;
  const size_t base = (size_t)b * 4096 * 1024 + h * 64;
  const __bf16* Kp = Kb + base + d0 + lr;  // column (h*64+d0+lr), index by t*1024

  floatx8 acc[4];
#pragma unroll
  for (int i = 0; i < 4; i++) acc[i] = zero8();
  floatx8 accs = zero8();  // Ksum accumulator (vs all-ones B)

  FragU ones;
#pragma unroll
  for (int j = 0; j < 16; j++) ones.e[j] = (__bf16)1.0f;

  const int tbeg = s * 512, tend = tbeg + 512;
  for (int t0 = tbeg; t0 < tend; t0 += 32) {
    FragU fa;  // A = K^T: A[m=d][k=t]
#pragma unroll
    for (int j = 0; j < 8; j++) {
      fa.e[j]     = Kp[(size_t)(t0 + half * 16 + j) * 1024];
      fa.e[8 + j] = Kp[(size_t)(t0 + half * 16 + 8 + j) * 1024];
    }
#pragma unroll
    for (int ni = 0; ni < 4; ni++) {
      const __bf16* Vp = Vb + base + ni * 16 + lr;  // B[k=t][n=e]
      FragU fb;
#pragma unroll
      for (int j = 0; j < 16; j++)
        fb.e[j] = Vp[(size_t)(t0 + half * 16 + j) * 1024];
      acc[ni] = wmma_bf16(fa.bf, fb.bf, acc[ni]);
    }
    accs = wmma_bf16(fa.bf, ones.bf, accs);  // column sums of K -> Ksum[d]
  }

  // store partial KV^T (fp32): kvp[(s*64+bh)][e][d]
  float* Kout = kvp + ((size_t)s * 64 + bh) * 4096;
#pragma unroll
  for (int ni = 0; ni < 4; ni++)
#pragma unroll
    for (int r = 0; r < 8; r++)
      Kout[(size_t)(ni * 16 + lr) * 64 + d0 + r + half * 8] = acc[ni][r];

  // store partial Ksum (fp32): every column n holds the same value; use n=0.
  if (lr == 0) {
    float* Sout = ksump + ((size_t)s * 64 + bh) * 64;
#pragma unroll
    for (int r = 0; r < 8; r++) Sout[d0 + r + half * 8] = accs[r];
  }
}

// ---------------------------------------------------------------------------
// KV aggregation stage 2: reduce 8 fp32 slabs -> bf16 KV^T + fp32 Ksum.
// grid = 64 (bh), WG = 256.
// ---------------------------------------------------------------------------
__global__ __launch_bounds__(256) void la_kv_reduce(
    const float* __restrict__ kvp, const float* __restrict__ ksump,
    unsigned short* __restrict__ kvt_raw, float* __restrict__ ksum) {
  __bf16* KVT = (__bf16*)kvt_raw;
  const int bh = blockIdx.x;
#pragma unroll
  for (int j = 0; j < 16; j++) {
    const int idx = threadIdx.x + j * 256;
    float s = 0.f;
#pragma unroll
    for (int sl = 0; sl < 8; sl++)
      s += kvp[((size_t)sl * 64 + bh) * 4096 + idx];
    KVT[(size_t)bh * 4096 + idx] = (__bf16)s;
  }
  if (threadIdx.x < 64) {
    float s = 0.f;
#pragma unroll
    for (int sl = 0; sl < 8; sl++)
      s += ksump[((size_t)sl * 64 + bh) * 64 + threadIdx.x];
    ksum[bh * 64 + threadIdx.x] = s;
  }
}

// ---------------------------------------------------------------------------
// Attention apply: O[t,e] = (Q[t,:] @ KV[:,e]) / (Q[t,:]·Ksum + 1e-6)
// grid.x = T/128 blocks, grid.y = b*h. WG 256 thr; wave owns 16 rows of t.
// ---------------------------------------------------------------------------
__global__ __launch_bounds__(256) void la_attn(
    const unsigned short* __restrict__ qb_raw,
    const unsigned short* __restrict__ kvt_raw,
    const float* __restrict__ ksum, unsigned short* __restrict__ ob_raw) {
  const __bf16* Qb = (const __bf16*)qb_raw;
  const __bf16* KVT = (const __bf16*)kvt_raw;
  __bf16* Ob = (__bf16*)ob_raw;
  __shared__ float zl[128];  // 8 waves x 16 rows

  const int bh = blockIdx.y, b = bh >> 4, h = bh & 15;
  const int wave = threadIdx.x >> 5, lane = threadIdx.x & 31;
  const int lr = lane & 15, half = lane >> 4;
  const int t0 = blockIdx.x * 128 + wave * 16;
  const size_t rowbase = (size_t)b * 4096 + t0;
  const __bf16* Arow = Qb + rowbase * 1024 + h * 64;
  const __bf16* Bw = KVT + (size_t)bh * 4096;  // KVT[e][d], ld=64

  floatx8 acc[4];
#pragma unroll
  for (int i = 0; i < 4; i++) acc[i] = zero8();

  bhalf16 a0 = load_fragA_row(Arow, 1024, 0, lr, half);
  bhalf16 a1 = load_fragA_row(Arow, 1024, 32, lr, half);
  bhalf16 b0[4], b1[4];
#pragma unroll
  for (int ni = 0; ni < 4; ni++) {
    b0[ni] = load_fragB_wt(Bw + (size_t)(ni * 16) * 64, 64, lr, half, 0);
    b1[ni] = load_fragB_wt(Bw + (size_t)(ni * 16) * 64, 64, lr, half, 32);
  }
#pragma unroll
  for (int ni = 0; ni < 4; ni++) acc[ni] = wmma_bf16(a0, b0[ni], acc[ni]);
#pragma unroll
  for (int ni = 0; ni < 4; ni++) acc[ni] = wmma_bf16(a1, b1[ni], acc[ni]);

  // normalizer z = 1 / (Q[row]·Ksum + 1e-6): each lane dots 32 of 64 elems,
  // halves combined with a wave32 xor-shuffle across the lane-16 boundary.
  {
    const __bf16* Qrow = Arow + (size_t)lr * 1024 + half * 32;
    const float* Ks = ksum + bh * 64 + half * 32;
    float p = 0.f;
#pragma unroll
    for (int j = 0; j < 32; j++) p += (float)Qrow[j] * Ks[j];
    p += __shfl_xor(p, 16);
    if (half == 0) zl[wave * 16 + lr] = 1.f / (p + 1e-6f);
  }
  __syncthreads();

#pragma unroll
  for (int ni = 0; ni < 4; ni++) {
#pragma unroll
    for (int r = 0; r < 8; r++) {
      const float zz = zl[wave * 16 + r + half * 8];
      Ob[(rowbase + r + half * 8) * 1024 + h * 64 + ni * 16 + lr] =
          (__bf16)(acc[ni][r] * zz);
    }
  }
}

// ---------------------------------------------------------------------------
// Output projection: out = O @ Wo^T + bo  (fp32 output). LDS-staged B tile.
// ---------------------------------------------------------------------------
__global__ __launch_bounds__(256) void la_gemm_out(
    const unsigned short* __restrict__ ob_raw,
    const unsigned short* __restrict__ wo_raw,
    const float* __restrict__ bo, float* __restrict__ out) {
  __shared__ __align__(16) unsigned short btile[2][64 * TSTR];  // 10 KB

  const __bf16* Ob = (const __bf16*)ob_raw;
  const __bf16* W = (const __bf16*)wo_raw;

  const int wave = threadIdx.x >> 5, lane = threadIdx.x & 31;
  const int lr = lane & 15, half = lane >> 4;
  const int m0 = blockIdx.x * 128 + wave * 16;
  const int n0 = blockIdx.y * 64;

  const int sn = threadIdx.x >> 2, sk = (threadIdx.x & 3) * 8;
  const __bf16* Wst = W + (size_t)(n0 + sn) * 1024 + sk;

  const __bf16* Arow = Ob + (size_t)m0 * 1024;
  floatx8 acc[4];
#pragma unroll
  for (int i = 0; i < 4; i++) acc[i] = zero8();

  *(uintx4*)&btile[0][sn * TSTR + sk] = *(const uintx4*)Wst;

  for (int it = 0; it < 32; ++it) {
    const int k0 = it * 32;
    const int cur = it & 1;
    uintx4 wreg;
    if (it + 1 < 32) wreg = *(const uintx4*)(Wst + k0 + 32);
    __syncthreads();
    bhalf16 a = load_fragA_row(Arow, 1024, k0, lr, half);
    bhalf16 bfr[4];
#pragma unroll
    for (int ni = 0; ni < 4; ni++)
      bfr[ni] = lds_fragB(&btile[cur][0], ni * 16 + lr, half);
#pragma unroll
    for (int ni = 0; ni < 4; ni++)
      acc[ni] = wmma_bf16(a, bfr[ni], acc[ni]);
    if (it + 1 < 32) *(uintx4*)&btile[cur ^ 1][sn * TSTR + sk] = wreg;
  }

#pragma unroll
  for (int ni = 0; ni < 4; ni++) {
    const int col = n0 + ni * 16 + lr;
    const float bb = bo[col];
#pragma unroll
    for (int r = 0; r < 8; r++)
      out[(size_t)(m0 + r + half * 8) * 1024 + col] = acc[ni][r] + bb;
  }
}

// ---------------------------------------------------------------------------
// Host launch
// ---------------------------------------------------------------------------
extern "C" void kernel_launch(void* const* d_in, const int* in_sizes, int n_in,
                              void* d_out, int out_size, void* d_ws,
                              size_t ws_size, hipStream_t stream) {
  const float* x  = (const float*)d_in[0];
  const float* Wq = (const float*)d_in[1];
  const float* bq = (const float*)d_in[2];
  const float* Wk = (const float*)d_in[3];
  const float* bk = (const float*)d_in[4];
  const float* Wv = (const float*)d_in[5];
  const float* bv = (const float*)d_in[6];
  const float* Wo = (const float*)d_in[7];
  const float* bo = (const float*)d_in[8];

  char* ws = (char*)d_ws;
  const size_t MB = 1ull << 20;
  unsigned short* xb    = (unsigned short*)(ws);                // 32 MB  x bf16
  unsigned short* wb    = (unsigned short*)(ws + 32 * MB);      // 8 MB   Wq,Wk,Wv,Wo bf16
  unsigned short* qb    = (unsigned short*)(ws + 40 * MB);      // 32 MB  Q bf16
  unsigned short* kb    = (unsigned short*)(ws + 72 * MB);      // 32 MB  K bf16
  unsigned short* vb    = (unsigned short*)(ws + 104 * MB);     // 32 MB  V bf16
  unsigned short* ob    = (unsigned short*)(ws + 136 * MB);     // 32 MB  O bf16
  unsigned short* kvt   = (unsigned short*)(ws + 168 * MB);     // 512 KB KV^T bf16
  float*          ksum  = (float*)(ws + 168 * MB + 512 * 1024); // 16 KB  Ksum f32
  float*          kvp   = (float*)(ws + 169 * MB);              // 8 MB   KV^T partials f32
  float*          ksump = (float*)(ws + 177 * MB);              // 128 KB Ksum partials f32

  const int NX = 4 * 4096 * 1024;  // 16777216
  const int NW = 1024 * 1024;

  la_cvt_bf16<<<NX / 512, 256, 0, stream>>>(x, xb, NX);
  la_cvt_bf16<<<NW / 512, 256, 0, stream>>>(Wq, wb + 0 * (size_t)NW, NW);
  la_cvt_bf16<<<NW / 512, 256, 0, stream>>>(Wk, wb + 1 * (size_t)NW, NW);
  la_cvt_bf16<<<NW / 512, 256, 0, stream>>>(Wv, wb + 2 * (size_t)NW, NW);
  la_cvt_bf16<<<NW / 512, 256, 0, stream>>>(Wo, wb + 3 * (size_t)NW, NW);

  la_gemm_qkv<<<dim3(128, 16, 3), 256, 0, stream>>>(xb, wb, bq, bk, bv, qb, kb, vb);
  la_kv_partial<<<dim3(64, 8), 128, 0, stream>>>(kb, vb, kvp, ksump);
  la_kv_reduce<<<64, 256, 0, stream>>>(kvp, ksump, kvt, ksum);
  la_attn<<<dim3(32, 64), 256, 0, stream>>>(qb, kvt, ksum, ob);
  la_gemm_out<<<dim3(128, 16), 256, 0, stream>>>(ob, wb + 3 * (size_t)NW, bo,
                                                 (float*)d_out);
}
